// ImplicitNetwork_33303176413211
// MI455X (gfx1250) — matrix operand order repaired and checked by reading, not compiled, not run on hardware
//
#include <hip/hip_runtime.h>
#include <hip/hip_bf16.h>
#include <math.h>

// ---------------------------------------------------------------------------
// MI455X (gfx1250) implementation.
//   Stage 1: ball query + feature build (particles staged to LDS with
//            GLOBAL_LOAD_ASYNC_TO_LDS_B128)
//   Stage 2: weight conversion f32 -> f16 padded (B-fragment friendly layout)
//   Stage 3: 9 x WMMA GEMM layers (v_wmma_f32_16x16x32_f16):
//            - B panel loaded by the Tensor Data Mover (tensor_load_to_lds)
//            - A tiles double-buffered via async-to-LDS (ASYNCcnt)
//            - fused bias + softplus(beta=100) epilogue, layer-3 skip concat,
//              final f32 store with row stride 257 (non-temporal).
// ---------------------------------------------------------------------------

typedef __attribute__((ext_vector_type(16))) _Float16 v16h;
typedef __attribute__((ext_vector_type(8)))  _Float16 v8h;
typedef __attribute__((ext_vector_type(8)))  float    v8f;
typedef __attribute__((ext_vector_type(4)))  unsigned int v4u;
typedef __attribute__((ext_vector_type(8)))  int      v8i;
typedef __attribute__((ext_vector_type(4)))  int      v4i;

union V16U { v16h v; v8h h[2]; };

#define N_PTS    32768      // 512 rays * 64 samples
#define N_PART   4096
#define KNN      20
#define RADIUS_F 0.05f
#define R2_F     (RADIUS_F * RADIUS_F)
#define KPAD     256        // padded K (input dim) for every layer
#define NPADW    320        // padded weight rows per layer (covers 257 -> 5*64)
#define BM       128
#define BN       64
#define NLAYERS  9
#define D_LAST   257
#define INV_SQRT2 0.70710678118654752f

// ---------------------------------------------------------------------------
// CDNA5 async/TDM helpers
// ---------------------------------------------------------------------------
__device__ __forceinline__ unsigned lds_addr_of(const void* p) {
  // Generic pointers into LDS carry the LDS byte offset in bits [31:0]
  // (ISA 10.2 aperture mapping).
  return (unsigned)(unsigned long long)(uintptr_t)p;
}

__device__ __forceinline__ void async_copy_b128(unsigned lds_off, const void* gptr) {
  asm volatile("global_load_async_to_lds_b128 %0, %1, off"
               :: "v"(lds_off), "v"((unsigned long long)(uintptr_t)gptr)
               : "memory");
}

__device__ __forceinline__ void wait_asynccnt0() {
  asm volatile("s_wait_asynccnt 0x0" ::: "memory");
}

// Issue one 2-D TDM descriptor: copy `rows` x `cols_halfs` f16 tile (row
// stride `stride_halfs` in global memory) into LDS at lds_byte_addr.
__device__ __forceinline__ void tdm_load_tile_f16(
    unsigned lds_byte_addr, const _Float16* gptr,
    int rows, int cols_halfs, int stride_halfs, int tensor_rows)
{
  unsigned long long ga = (unsigned long long)(uintptr_t)gptr;
  // D# group 0: count=1, lds_addr, global tile addr, type=2 ("image")
  v4u g0 = { 1u,
             lds_byte_addr,
             (unsigned)(ga & 0xFFFFFFFFu),
             (unsigned)((ga >> 32) & 0x01FFFFFFu) | (2u << 30) };
  // D# group 1 (bit layout per ISA 8.4):
  //  w0: workgroup_mask=0, data_size=1 (2B) at [17:16]
  //  w1: [47:32]=atomic_barrier_addr=0, [63:48]=tensor_dim0[15:0]
  //  w2: [79:64]=tensor_dim0[31:16]=0, [95:80]=tensor_dim1[15:0]
  //  w3: [111:96]=tensor_dim1[31:16]=0, [127:112]=tile_dim0
  //  w4: [143:128]=tile_dim1, [159:144]=tile_dim2=0
  //  w5: tensor_dim0_stride[31:0]
  //  w6: [207:192]=stride[47:32]=0, [223:208]=tensor_dim1_stride=0
  v8i g1 = { (int)(1u << 16),
             (int)((unsigned)cols_halfs << 16),
             (int)((unsigned)tensor_rows << 16),
             (int)((unsigned)cols_halfs << 16),
             rows,
             stride_halfs,
             0, 0 };
  v4i g2 = { 0, 0, 0, 0 };
  v4i g3 = { 0, 0, 0, 0 };
#if __has_include(<hip/amd_detail/amd_gfx1250_TDM.h>)
  v8i g4 = { 0, 0, 0, 0, 0, 0, 0, 0 };
  __builtin_amdgcn_tensor_load_to_lds(g0, g1, g2, g3, g4, 0);
#else
  __builtin_amdgcn_tensor_load_to_lds(g0, g1, g2, g3, 0);
#endif
}

// ---------------------------------------------------------------------------
// math helpers
// ---------------------------------------------------------------------------
__device__ __forceinline__ float softplus100(float x) {
  float t = 100.0f * x;
  return (t > 20.0f) ? x : log1pf(__expf(t)) * 0.01f;
}

__device__ __forceinline__ int emb3(_Float16* f, int o, float x0, float x1, float x2) {
  f[o + 0] = (_Float16)x0; f[o + 1] = (_Float16)x1; f[o + 2] = (_Float16)x2;
  o += 3;
  float s = 1.0f;
#pragma unroll
  for (int fr = 0; fr < 10; ++fr) {
    f[o + 0] = (_Float16)sinf(x0 * s);
    f[o + 1] = (_Float16)sinf(x1 * s);
    f[o + 2] = (_Float16)sinf(x2 * s);
    f[o + 3] = (_Float16)cosf(x0 * s);
    f[o + 4] = (_Float16)cosf(x1 * s);
    f[o + 5] = (_Float16)cosf(x2 * s);
    o += 6; s *= 2.0f;
  }
  return o;
}

__device__ __forceinline__ int emb1(_Float16* f, int o, float x) {
  f[o++] = (_Float16)x;
  float s = 1.0f;
#pragma unroll
  for (int fr = 0; fr < 4; ++fr) {
    f[o++] = (_Float16)sinf(x * s);
    f[o++] = (_Float16)cosf(x * s);
    s *= 2.0f;
  }
  return o;
}

// ---------------------------------------------------------------------------
// Stage 1: ball query + features.  Particles async-staged into LDS; one
// thread per query point; two passes over the particle list.
// ---------------------------------------------------------------------------
__global__ __launch_bounds__(256) void feat_kernel(
    const float* __restrict__ inp, const float* __restrict__ part,
    _Float16* __restrict__ feat, float* __restrict__ dists,
    float* __restrict__ numnn)
{
  __shared__ float Ps[N_PART * 3];                 // 48 KB of the 320 KB pool
  // 3072 x 16B chunks, 12 per thread, DMA'd LDS-direct (no VGPR bounce)
  for (int c = threadIdx.x; c < (N_PART * 3) / 4; c += 256)
    async_copy_b128(lds_addr_of(&Ps[c * 4]), part + c * 4);
  wait_asynccnt0();
  __syncthreads();

  const int i = blockIdx.x * blockDim.x + threadIdx.x;
  const float qx = inp[i * 3 + 0], qy = inp[i * 3 + 1], qz = inp[i * 3 + 2];
  float* dout = dists + (size_t)i * KNN;
  const float rinv = 1.0f / RADIUS_F;

  float sw = 0.f, swx = 0.f, swy = 0.f, swz = 0.f;   // kernel-weighted sums
  float svx = 0.f, svy = 0.f, svz = 0.f;             // offset sums (for mean)
  int cnt = 0;
  for (int p = 0; p < N_PART && cnt < KNN; ++p) {
    float px = Ps[p * 3 + 0], py = Ps[p * 3 + 1], pz = Ps[p * 3 + 2];
    float dx = px - qx, dy = py - qy, dz = pz - qz;
    float d2 = dx * dx + dy * dy + dz * dz;
    if (d2 < R2_F) {
      __builtin_nontemporal_store(d2, &dout[cnt]);
      float d = sqrtf(d2);
      float u = d * rinv;
      float w = fmaxf(1.0f - u * u * u, 0.0f);
      sw += w; swx += w * px; swy += w * py; swz += w * pz;
      svx += dx; svy += dy; svz += dz;
      ++cnt;
    }
  }
  for (int s = cnt; s < KNN; ++s) __builtin_nontemporal_store(0.0f, &dout[s]);

  // invalid slots contribute w computed from the zero-neighbor (ref semantics)
  float dq = sqrtf(qx * qx + qy * qy + qz * qz);
  float u0 = dq * rinv;
  float w0 = fmaxf(1.0f - u0 * u0 * u0, 0.0f);
  sw += (float)(KNN - cnt) * w0;

  const float fnum = (float)cnt;
  numnn[i] = fnum;
  const float invn = 1.0f / (fnum + 1e-12f);
  const float mx = svx * invn, my = svy * invn, mz = svz * invn;
  const float invw = 1.0f / (sw + 1e-12f);
  const float smx = swx * invw, smy = swy * invw, smz = swz * invw;

  // pass 2: masked variance of neighbor offsets
  float vx = 0.f, vy = 0.f, vz = 0.f;
  int c2 = 0;
  for (int p = 0; p < N_PART && c2 < KNN; ++p) {
    float dx = Ps[p * 3 + 0] - qx, dy = Ps[p * 3 + 1] - qy, dz = Ps[p * 3 + 2] - qz;
    float d2 = dx * dx + dy * dy + dz * dz;
    if (d2 < R2_F) {
      float ex = dx - mx, ey = dy - my, ez = dz - mz;
      vx += ex * ex; vy += ey * ey; vz += ez * ez;
      ++c2;
    }
  }
  vx *= invn; vy *= invn; vz *= invn;

  _Float16* f = feat + (size_t)i * KPAD;
  int o = 0;
  o = emb3(f, o, qx, qy, qz);        // 63
  o = emb1(f, o, sw);                // 9  (density)
  o = emb3(f, o, smx, smy, smz);     // 63 (smoothed position)
  o = emb3(f, o, vx, vy, vz);        // 63 (variance)  -> 198 total
  for (; o < KPAD; ++o) f[o] = (_Float16)0.0f;  // zero pad to K=256
}

// ---------------------------------------------------------------------------
// Stage 2: pack weights.  Wp[l][n][k] f16, row n = output neuron so that
// WMMA B fragments are two contiguous 16-byte LDS reads per lane (no transpose needed).
// ---------------------------------------------------------------------------
struct LayerArgs {
  const float* W[NLAYERS];
  const float* B[NLAYERS];
};

__global__ __launch_bounds__(256) void prep_kernel(
    LayerArgs la, _Float16* __restrict__ wp, float* __restrict__ bp)
{
  const int DO[NLAYERS] = {256, 256, 256, 253, 256, 256, 256, 256, 257};
  const int DI[NLAYERS] = {198, 256, 256, 256, 256, 256, 256, 256, 256};
  const int l = blockIdx.y;
  const int n = blockIdx.x;        // 0..NPADW-1
  const int k = threadIdx.x;       // 0..255
  const int d_o = DO[l], d_i = DI[l];
  float v = (n < d_o && k < d_i) ? la.W[l][n * d_i + k] : 0.0f;
  wp[((size_t)l * NPADW + n) * KPAD + k] = (_Float16)v;
  if (k == 0) bp[l * NPADW + n] = (n < d_o) ? la.B[l][n] : 0.0f;
}

// ---------------------------------------------------------------------------
// Stage 3: WMMA GEMM.  Block = 256 threads = 8 wave32s in a 4(M) x 2(N) grid;
// block tile 128x64, wave tile 32x32 = 2x2 v_wmma_f32_16x16x32_f16 tiles.
// B panel (64x256 f16 = 32 KB): one TDM descriptor issued by wave 0.
// A tiles (128x32 f16 = 8 KB): double-buffered async-to-LDS.
//   mode 0: softplus -> f16 activations
//   mode 1: softplus * inv_sqrt2, cols>=253 <- hit*inv_sqrt2 (layer 3)
//   mode 2: final layer, raw f32 to Xout (col < 257), non-temporal
// ---------------------------------------------------------------------------
__global__ __launch_bounds__(256) void mlp_gemm(
    const _Float16* __restrict__ A, const _Float16* __restrict__ W,
    const float* __restrict__ bias, _Float16* __restrict__ Aout,
    float* __restrict__ Xout, const float* __restrict__ hit, int mode)
{
  __shared__ _Float16 Bs[BN * KPAD];      // 32 KB
  __shared__ _Float16 As[2][BM * 32];     // 2 x 8 KB (double buffer)

  const int t  = threadIdx.x;
  const int m0 = blockIdx.x * BM;
  const int n0 = blockIdx.y * BN;

  // --- B panel: single TDM 2-D tile DMA (wave 0 issues; TENSORcnt-tracked)
  if (t < 32) {
    tdm_load_tile_f16(lds_addr_of(&Bs[0]), W + (size_t)n0 * KPAD,
                      /*rows=*/BN, /*cols_halfs=*/KPAD,
                      /*stride_halfs=*/KPAD, /*tensor_rows=*/NPADW);
    __builtin_amdgcn_s_wait_tensorcnt(0);
  }

  // --- A tile staging: 512 x 16B chunks, 2 per thread, async-to-LDS
  auto issueA = [&](int kt, int buf) {
#pragma unroll
    for (int i = 0; i < 2; ++i) {
      int c  = t + i * 256;
      int r  = c >> 2;
      int ko = (c & 3) * 8;
      async_copy_b128(lds_addr_of(&As[buf][r * 32 + ko]),
                      &A[((size_t)(m0 + r)) * KPAD + kt * 32 + ko]);
    }
  };

  issueA(0, 0);
  wait_asynccnt0();
  __syncthreads();          // Bs + As[0] visible to all waves

  const int wave = t >> 5, lane = t & 31;
  const int wm = wave >> 1, wn = wave & 1;
  const int lx = lane & 15, lg = lane >> 4;

  v8f acc[2][2] = {};

  for (int kt = 0; kt < KPAD / 32; ++kt) {
    const int buf = kt & 1;
    if (kt + 1 < KPAD / 32) issueA(kt + 1, buf ^ 1);   // overlap DMA w/ math

    // A fragments (ISA 7.12.2: lanes 0-15 / 16-31 both hold M=0..15; the lane
    // group selects which 8-K runs the VGPR pairs carry)
    v16h af[2], bf[2];
#pragma unroll
    for (int tm = 0; tm < 2; ++tm) {
      V16U u;
      const _Float16* p = &As[buf][(wm * 32 + tm * 16 + lx) * 32];
      u.h[0] = *(const v8h*)(p + lg * 8);
      u.h[1] = *(const v8h*)(p + 16 + lg * 8);
      af[tm] = u.v;
    }
    // B fragments: lane = N column, halves = 8-K runs at lg*8 and 16+lg*8
#pragma unroll
    for (int tn = 0; tn < 2; ++tn) {
      V16U u;
      const _Float16* p = &Bs[(wn * 32 + tn * 16 + lx) * KPAD + kt * 32 + lg * 8];
      u.h[0] = *(const v8h*)(p);
      u.h[1] = *(const v8h*)(p + 16);
      bf[tn] = u.v;
    }

#pragma unroll
    for (int tm = 0; tm < 2; ++tm)
#pragma unroll
      for (int tn = 0; tn < 2; ++tn)
        acc[tm][tn] = __builtin_amdgcn_wmma_f32_16x16x32_f16(
            false, af[tm], false, bf[tn], (short)0, acc[tm][tn], false, false);

    if (kt + 1 < KPAD / 32) {
      wait_asynccnt0();     // next buffer's DMA landed
      __syncthreads();      // and everyone is done reading the current one
    }
  }

  // Epilogue.  C/D layout: lane -> N = lane&15, M = (lane>>4)*8 + r
#pragma unroll
  for (int tm = 0; tm < 2; ++tm) {
#pragma unroll
    for (int tn = 0; tn < 2; ++tn) {
      const int col = n0 + wn * 32 + tn * 16 + lx;
      const float b = bias[col];
#pragma unroll
      for (int r = 0; r < 8; ++r) {
        const int row = m0 + wm * 32 + tm * 16 + lg * 8 + r;
        float v = acc[tm][tn][r] + b;
        if (mode == 2) {
          if (col < D_LAST)
            __builtin_nontemporal_store(v, &Xout[(size_t)row * D_LAST + col]);
        } else {
          float a = softplus100(v);
          if (mode == 1) {
            a *= INV_SQRT2;
            if (col >= 253) a = hit[row * 3 + (col - 253)] * INV_SQRT2;
          }
          Aout[(size_t)row * KPAD + col] = (_Float16)a;
        }
      }
    }
  }
}

// ---------------------------------------------------------------------------
// Host side.  Workspace layout (bytes):
//   [0, 16M)        actA  : f16 activations ping  (32768 x 256)
//   [16M, 32M)      actB  : f16 activations pong
//   [32M, +1.44M)   wp    : f16 packed weights, 9 x 320 x 256
//   then            bp    : f32 padded biases, 9 x 320
// ---------------------------------------------------------------------------
extern "C" void kernel_launch(void* const* d_in, const int* in_sizes, int n_in,
                              void* d_out, int out_size, void* d_ws, size_t ws_size,
                              hipStream_t stream) {
  (void)in_sizes; (void)n_in; (void)out_size; (void)ws_size;

  const float* inp  = (const float*)d_in[0];   // [512,64,3]
  const float* part = (const float*)d_in[1];   // [4096,3]
  LayerArgs la;
  for (int l = 0; l < NLAYERS; ++l) {
    la.W[l] = (const float*)d_in[2 + 2 * l];
    la.B[l] = (const float*)d_in[3 + 2 * l];
  }

  const size_t ACT_BYTES = (size_t)N_PTS * KPAD * sizeof(_Float16);   // 16 MB
  const size_t WP_BYTES  = (size_t)NLAYERS * NPADW * KPAD * sizeof(_Float16);
  char* ws = (char*)d_ws;
  _Float16* actA = (_Float16*)ws;
  _Float16* actB = (_Float16*)(ws + ACT_BYTES);
  _Float16* wp   = (_Float16*)(ws + 2 * ACT_BYTES);
  float*    bp   = (float*)(ws + 2 * ACT_BYTES + WP_BYTES);

  float* xout  = (float*)d_out;                       // [32768, 257]
  float* dists = xout + (size_t)N_PTS * D_LAST;       // [32768, 20]
  float* numnn = dists + (size_t)N_PTS * KNN;         // [32768]

  // Stage 2 (independent of stage 1; both on stream, serialized is fine)
  prep_kernel<<<dim3(NPADW, NLAYERS), 256, 0, stream>>>(la, wp, bp);

  // Stage 1
  feat_kernel<<<dim3(N_PTS / 256), 256, 0, stream>>>(inp, part, actA, dists, numnn);

  // Stage 3: nine GEMM layers, ping-pong activations
  for (int l = 0; l < NLAYERS; ++l) {
    const _Float16* Ain = (l % 2 == 0) ? actA : actB;
    _Float16* Aout      = (l % 2 == 0) ? actB : actA;
    int mode = (l == NLAYERS - 1) ? 2 : (l == 3 ? 1 : 0);
    int ntn  = (l == NLAYERS - 1) ? 5 : 4;   // 257 -> 5 x 64-col tiles
    mlp_gemm<<<dim3(N_PTS / BM, ntn), 256, 0, stream>>>(
        Ain, wp + (size_t)l * NPADW * KPAD, bp + l * NPADW,
        Aout, xout, inp, mode);
  }
}
